// LSTMModel_69054484185610
// MI455X (gfx1250) — matrix operand order repaired and checked by reading, not compiled, not run on hardware
//
#include <hip/hip_runtime.h>
#include <hip/hip_bf16.h>

// ---------------------------------------------------------------------------
// LSTM forward on MI455X (gfx1250), wave32 + V_WMMA_F32_16X16X32_BF16.
//   z = [x_t, h] @ [Wx; Wh] + b   (fused K = 256 + 1024 = 1280)
// Software-pipelined K loop with prefetch distance 2 and explicit
// sched_barriers so the scheduler cannot collapse the pipeline; all
// addresses affine so loads fold into immediate offsets.
// ---------------------------------------------------------------------------

typedef __bf16 bf16_t;
typedef __attribute__((ext_vector_type(16))) __bf16 v16bf;
typedef __attribute__((ext_vector_type(8)))  float  v8f;
typedef __attribute__((ext_vector_type(4)))  unsigned int u32x4;

#define UNITS 1024
#define BATCH 128
#define SEQ   512
#define DIN   256
#define KDIM  (DIN + UNITS)   // 1280
#define GATES (4 * UNITS)     // 4096

// f32 -> bf16, round-to-nearest-even
__device__ __forceinline__ bf16_t f2bf(float f) {
  union { float f; unsigned u; } in; in.f = f;
  unsigned u = in.u;
  unsigned r = u + 0x7FFFu + ((u >> 16) & 1u);
  union { unsigned short s; bf16_t b; } out;
  out.s = (unsigned short)(r >> 16);
  return out.b;
}

// Branch-free tanh/sigmoid on v_exp_f32 (correct limits at +/-inf).
__device__ __forceinline__ float fast_tanh(float x) {
  return 1.0f - 2.0f / (__expf(2.0f * x) + 1.0f);
}
__device__ __forceinline__ float fast_sigmoid(float x) {
  return 1.0f / (1.0f + __expf(-x));
}

// Build one 16-element bf16 WMMA fragment from two 16-byte chunks.
__device__ __forceinline__ v16bf load_frag2(const bf16_t* p0, const bf16_t* p1) {
  union { u32x4 u[2]; v16bf v; } fr;
  fr.u[0] = *(const u32x4*)p0;
  fr.u[1] = *(const u32x4*)p1;
  return fr.v;
}

// A fragment (16-bit A 16x32 layout): lane holds row (lane&15);
// K = {akoff..akoff+7} U {16+akoff..16+akoff+7}, akoff = 8*(lane>=16).
__device__ __forceinline__ v16bf loadA(const bf16_t* arp, int akoff) {
  return load_frag2(arp + akoff, arp + 16 + akoff);
}

// B fragments for the 4 gates. wrp already includes column row-base and the
// per-lane K offset (8*(lane>=16)); gates are UNITS rows apart in Wt.
__device__ __forceinline__ void loadB4(v16bf b[4], const bf16_t* wrp) {
#pragma unroll
  for (int g = 0; g < 4; ++g) {
    const bf16_t* p = wrp + (size_t)g * ((size_t)UNITS * KDIM);
    b[g] = load_frag2(p, p + 16);
  }
}

__device__ __forceinline__ void wmma4(v8f acc[4], v16bf a, const v16bf b[4]) {
#pragma unroll
  for (int g = 0; g < 4; ++g)
    acc[g] = __builtin_amdgcn_wmma_f32_16x16x32_bf16(
        false, a, false, b[g], (short)0, acc[g], false, false);
}

// Pipelined GEMM phase over NC chunks of K=32. Prefetch distance 2 with a
// 3-deep rotating fragment buffer; sched_barrier(0) pins load issue ahead of
// the consuming WMMAs so waitcnt insertion emits partial waits.
template <int NC>
__device__ __forceinline__ void gemm_phase(v8f acc[4], const bf16_t* arow,
                                           const bf16_t* wrow, int akoff) {
  constexpr int PF = (NC >= 3) ? 2 : 1;  // prefetch distance
  v16bf a[3];
  v16bf b[3][4];
#pragma unroll
  for (int s = 0; s < PF; ++s) {
    a[s] = loadA(arow + 32 * s, akoff);
    loadB4(b[s], wrow + 32 * s);
  }
  __builtin_amdgcn_sched_barrier(0);
#pragma unroll
  for (int c = 0; c < NC; ++c) {
    const int cur = c % 3;               // constant after full unroll
    if (c + PF < NC) {
      const int nxt = (c + PF) % 3;
      a[nxt] = loadA(arow + 32 * (c + PF), akoff);
      loadB4(b[nxt], wrow + 32 * (c + PF));
    }
    __builtin_amdgcn_sched_barrier(0);   // loads stay ahead of these WMMAs
    wmma4(acc, a[cur], b[cur]);
    __builtin_amdgcn_sched_barrier(0);   // WMMAs don't float up past loads
  }
}

// ---------------------------------------------------------------------------
// Prep kernels
// ---------------------------------------------------------------------------
__global__ void convert_x_kernel(const float* __restrict__ x,
                                 bf16_t* __restrict__ xb, int n) {
  int i = blockIdx.x * blockDim.x + threadIdx.x;
  if (i < n) xb[i] = f2bf(x[i]);
}

// Wt[n][k] = (k < DIN ? Wx[k][n] : Wh[k-DIN][n]) as bf16   (N-major, K contig)
__global__ void build_weights_kernel(const float* __restrict__ Wx,
                                     const float* __restrict__ Wh,
                                     bf16_t* __restrict__ Wt) {
  int i = blockIdx.x * blockDim.x + threadIdx.x;
  if (i >= GATES * KDIM) return;
  int n = i / KDIM;
  int k = i - n * KDIM;
  float v = (k < DIN) ? Wx[(size_t)k * GATES + n]
                      : Wh[(size_t)(k - DIN) * GATES + n];
  Wt[i] = f2bf(v);
}

__global__ void init_state_kernel(bf16_t* __restrict__ h0,
                                  float* __restrict__ cbuf) {
  int i = blockIdx.x * blockDim.x + threadIdx.x;
  if (i < BATCH * UNITS) { h0[i] = f2bf(0.0f); cbuf[i] = 0.0f; }
}

// ---------------------------------------------------------------------------
// One LSTM timestep: 512 tiles of 16x16 (batch x units); one wave per tile,
// computing all four gates (i,f,g,o) with shared A fragments.
// grid = 128 blocks x 128 threads (4 waves/block).
// ---------------------------------------------------------------------------
__global__ void __launch_bounds__(128)
lstm_step_kernel(const bf16_t* __restrict__ xb,    // [B][T][DIN] bf16
                 const bf16_t* __restrict__ Wt,    // [4096][1280] bf16
                 const float*  __restrict__ bias,  // [4096]
                 const bf16_t* __restrict__ h_in,  // [B][U] bf16
                 bf16_t*       __restrict__ h_out, // [B][U] bf16
                 float*        __restrict__ cbuf,  // [B][U] f32
                 float*        __restrict__ h_f32, // [B][U] f32
                 int t) {
  const int lane  = threadIdx.x & 31;
  const int wave  = threadIdx.x >> 5;
  const int tile  = blockIdx.x * 4 + wave;   // 0..511
  const int mbase = (tile >> 6) * 16;        // batch tile base (0..112)
  const int nbase = (tile & 63) * 16;        // unit tile base  (0..1008)

  const int l15 = lane & 15;
  const int hi  = lane >> 4;                 // 0 | 1
  const int arow = mbase + l15;              // A-matrix row (batch index)
  const int bcol = nbase + l15;              // B-matrix col (unit index)

  // Accumulators seeded with bias (acts as the C operand).
  v8f acc[4];
#pragma unroll
  for (int g = 0; g < 4; ++g) {
    float bv = bias[g * UNITS + bcol];
#pragma unroll
    for (int e = 0; e < 8; ++e) acc[g][e] = bv;
  }

  // Prefetch cell state early; latency hides under the GEMM.
  float cold[8];
#pragma unroll
  for (int r = 0; r < 8; ++r) {
    const int m = mbase + r + hi * 8;
    cold[r] = cbuf[(size_t)m * UNITS + bcol];
  }

  const bf16_t* xrow  = xb + ((size_t)arow * SEQ + t) * DIN;
  const bf16_t* hrow  = h_in + (size_t)arow * UNITS;
  const bf16_t* wbase = Wt + (size_t)bcol * KDIM + hi * 8;
  const int akoff = hi * 8;

  gemm_phase<DIN / 32>(acc, xrow, wbase, akoff);          // x_t @ Wx  (8 chunks)
  gemm_phase<UNITS / 32>(acc, hrow, wbase + DIN, akoff);  // h   @ Wh  (32 chunks)

  // Cell update directly from the C/D VGPR layout:
  //   vgpr r, lane l  ->  m = mbase + r + 8*(l>=16), n = nbase + (l&15)
#pragma unroll
  for (int r = 0; r < 8; ++r) {
    const int m = mbase + r + hi * 8;
    const size_t idx = (size_t)m * UNITS + bcol;
    float iv = fast_sigmoid(acc[0][r]);
    float fv = fast_sigmoid(acc[1][r]);
    float gv = fast_tanh(acc[2][r]);
    float ov = fast_sigmoid(acc[3][r]);
    float cn = fv * cold[r] + iv * gv;
    float hn = ov * fast_tanh(cn);
    cbuf[idx]  = cn;
    h_f32[idx] = hn;
    h_out[idx] = f2bf(hn);
  }
}

// ---------------------------------------------------------------------------
// logits = h_last @ Wout + bout; 2-way softmax. One thread per batch row.
// ---------------------------------------------------------------------------
__global__ void final_kernel(const float* __restrict__ h,
                             const float* __restrict__ Wout,
                             const float* __restrict__ bout,
                             float* __restrict__ out) {
  int b = threadIdx.x;
  if (b >= BATCH) return;
  float l0 = bout[0], l1 = bout[1];
  const float* hb = h + (size_t)b * UNITS;
  for (int u = 0; u < UNITS; ++u) {
    float hv = hb[u];
    l0 = fmaf(hv, Wout[2 * u + 0], l0);
    l1 = fmaf(hv, Wout[2 * u + 1], l1);
  }
  float mx = fmaxf(l0, l1);
  float e0 = __expf(l0 - mx), e1 = __expf(l1 - mx);
  float inv = 1.0f / (e0 + e1);
  out[2 * b + 0] = e0 * inv;
  out[2 * b + 1] = e1 * inv;
}

// ---------------------------------------------------------------------------
extern "C" void kernel_launch(void* const* d_in, const int* in_sizes, int n_in,
                              void* d_out, int out_size, void* d_ws,
                              size_t ws_size, hipStream_t stream) {
  (void)in_sizes; (void)n_in; (void)out_size; (void)ws_size;

  const float* data = (const float*)d_in[0];  // [128,512,256]
  const float* Wx   = (const float*)d_in[1];  // [256,4096]
  const float* Wh   = (const float*)d_in[2];  // [1024,4096]
  const float* bias = (const float*)d_in[3];  // [4096]
  const float* Wout = (const float*)d_in[4];  // [1024,2]
  const float* bout = (const float*)d_in[5];  // [2]

  // Workspace carve-up (~44 MB total).
  char* ws = (char*)d_ws;
  size_t off = 0;
  bf16_t* xb = (bf16_t*)(ws + off); off += (size_t)BATCH * SEQ * DIN * 2; // 32 MB
  bf16_t* Wt = (bf16_t*)(ws + off); off += (size_t)GATES * KDIM * 2;     // 10 MB
  bf16_t* hA = (bf16_t*)(ws + off); off += (size_t)BATCH * UNITS * 2;
  bf16_t* hB = (bf16_t*)(ws + off); off += (size_t)BATCH * UNITS * 2;
  float*  cb = (float*)(ws + off);  off += (size_t)BATCH * UNITS * 4;
  float*  hf = (float*)(ws + off);  off += (size_t)BATCH * UNITS * 4;

  const int nx = BATCH * SEQ * DIN;
  convert_x_kernel<<<(nx + 255) / 256, 256, 0, stream>>>(data, xb, nx);
  const int nw = GATES * KDIM;
  build_weights_kernel<<<(nw + 255) / 256, 256, 0, stream>>>(Wx, Wh, Wt);
  init_state_kernel<<<(BATCH * UNITS + 255) / 256, 256, 0, stream>>>(hA, cb);

  for (int t = 0; t < SEQ; ++t) {
    const bf16_t* hin = (t & 1) ? hB : hA;
    bf16_t* hout      = (t & 1) ? hA : hB;
    lstm_step_kernel<<<128, 128, 0, stream>>>(xb, Wt, bias, hin, hout, cb, hf, t);
  }

  final_kernel<<<1, 128, 0, stream>>>(hf, Wout, bout, (float*)d_out);
}
